// PatchMatch_67430986547243
// MI455X (gfx1250) — matrix operand order repaired and checked by reading, not compiled, not run on hardware
//
#include <hip/hip_runtime.h>
#include <hip/hip_bf16.h>

typedef __bf16 bf16_t;
typedef __attribute__((ext_vector_type(16))) __bf16 v16bf;
typedef __attribute__((ext_vector_type(8)))  __bf16 v8bf;
typedef __attribute__((ext_vector_type(8)))  float   v8f;

#define HW   128
#define NCH  192
#define NPIX (HW * HW)

// ---------------------------------------------------------------------------
// helpers
// ---------------------------------------------------------------------------
__device__ __forceinline__ v16bf combine8(v8bf lo, v8bf hi) {
  v16bf r;
#pragma unroll
  for (int i = 0; i < 8; ++i) { r[i] = lo[i]; r[i + 8] = hi[i]; }
  return r;
}

// ---------------------------------------------------------------------------
// 1) repack fp32 [c][h][w] -> bf16 [h*w][c]  (pixel-major, RNE rounding)
// ---------------------------------------------------------------------------
__global__ __launch_bounds__(256) void pack_kernel(const float* __restrict__ src,
                                                   bf16_t* __restrict__ dst) {
  int t = blockIdx.x * blockDim.x + threadIdx.x;     // t < NCH*NPIX
  int c = t >> 14;                                   // / 16384
  int p = t & (NPIX - 1);
  unsigned u = __float_as_uint(src[t]);
  unsigned r = (u + 0x7FFFu + ((u >> 16) & 1u)) >> 16;   // round-nearest-even
  dst[(size_t)p * NCH + c] = __builtin_bit_cast(bf16_t, (unsigned short)r);
}

// ---------------------------------------------------------------------------
// 2) float coord maps -> int working buffers
// ---------------------------------------------------------------------------
__global__ __launch_bounds__(256) void init_kernel(const float* __restrict__ fx,
                                                   const float* __restrict__ fy,
                                                   int* __restrict__ cx,
                                                   int* __restrict__ cy) {
  int p = blockIdx.x * blockDim.x + threadIdx.x;
  cx[p] = (int)fx[p];
  cy[p] = (int)fy[p];
}

// ---------------------------------------------------------------------------
// 3) evaluate step: MODE 0 = propagation (reflect-shift neighbors),
//                   MODE 1 = random-search (jittered ring, mod 128)
// one wave32 handles one 16-pixel row tile; scores via v_wmma bf16 diagonal
// ---------------------------------------------------------------------------
template <int MODE>
__global__ __launch_bounds__(256) void eval_kernel(
    const bf16_t* __restrict__ inB, const bf16_t* __restrict__ refB,
    const int* __restrict__ curX, const int* __restrict__ curY,
    int* __restrict__ nxtX, int* __restrict__ nxtY, int k, int m) {
  const int lane  = threadIdx.x & 31;
  const int wave  = threadIdx.x >> 5;      // 8 waves per block = one image row
  const int half  = lane >> 4;             // K-chunk half select
  const int laneM = lane & 15;             // tile-local pixel index
  const int y     = blockIdx.x;
  const int px    = wave * 16 + laneM;
  const int p     = y * HW + px;

  // B fragments: input vectors of the 16-pixel tile (reused for all 9 cands)
  // B 32x16 bf16 layout: lane n -> col n, K {0..7,16..23} ; lane n+16 -> col n, K {8..15,24..31}
  const bf16_t* bbase = inB + (size_t)p * NCH + half * 8;
  v16bf Bf[6];
#pragma unroll
  for (int t = 0; t < 6; ++t) {
    const bf16_t* q = bbase + t * 32;
    Bf[t] = combine8(*(const v8bf*)q, *(const v8bf*)(q + 16));
  }

  float best   = -3.0e38f;
  int   bestAx = 0, bestAy = 0;

  const int DXtab[9] = {0, -1, -1, 0, 0, 0, 1, 1, 0};
  const int DYtab[9] = {1, 0, 0, 1, 0, -1, 0, 0, -1};

#pragma unroll
  for (int cnd = 0; cnd < 9; ++cnd) {
    const int di = cnd / 3 - 1;   // y offset
    const int dj = cnd % 3 - 1;   // x offset
    int ax, ay;
    if (MODE == 0) {
      // reflect-padded shift by (k*di, k*dj), gather neighbor's assignment
      int sy = y + k * di;
      sy = sy < 0 ? -sy : sy;
      sy = sy > HW - 1 ? 2 * (HW - 1) - sy : sy;
      int sx = px + k * dj;
      sx = sx < 0 ? -sx : sx;
      sx = sx > HW - 1 ? 2 * (HW - 1) - sx : sx;
      const int sp = sy * HW + sx;
      ax = curX[sp];
      ay = curY[sp];
    } else {
      // jittered candidates around own assignment, wrap mod 128
      const int x0 = curX[p], y0 = curY[p];
      ax = (x0 + k * dj + m * DXtab[cnd] + HW) & (HW - 1);
      ay = (y0 + k * di + m * DYtab[cnd] + HW) & (HW - 1);
    }
    const int qp = ay * HW + ax;

    // A 16x32 bf16 layout: lane L -> row L, K {0..7,16..23};
    //                      lane L+16 -> row L, K {8..15,24..31}
    const bf16_t* abase = refB + (size_t)qp * NCH + half * 8;
    v8f acc = {};
#pragma unroll
    for (int t = 0; t < 6; ++t) {
      const bf16_t* ap = abase + t * 32;
      v16bf A = combine8(*(const v8bf*)ap, *(const v8bf*)(ap + 16));
      acc = __builtin_amdgcn_wmma_f32_16x16x32_bf16(
          /*neg_a=*/false, A, /*neg_b=*/false, Bf[t],
          /*c_mod=*/(short)0, acc, /*reuse_a=*/false, /*reuse_b=*/false);
    }

    // diagonal element (p_local, p_local):
    //   p<8  -> VGPR p,   lane p        (lanes 0..7)
    //   p>=8 -> VGPR p-8, lane p+16     (lanes 24..31)
    const int sel = (lane < 8) ? lane : (lane - 24);
    float s = acc[0];
#pragma unroll
    for (int i = 1; i < 8; ++i) s = (sel == i) ? acc[i] : s;

    // branchless argmax (strict > keeps earliest candidate, matches argmax)
    const bool upd = s > best;
    best   = upd ? s : best;
    bestAx = upd ? ax : bestAx;
    bestAy = upd ? ay : bestAy;
  }

  if (lane < 8 || lane >= 24) {
    const int outp = y * HW + wave * 16 + ((lane < 8) ? lane : lane - 16);
    nxtX[outp] = bestAx;
    nxtY[outp] = bestAy;
  }
}

// ---------------------------------------------------------------------------
// 4) final: exact fp32 score + linear index, straight from original tensors
// ---------------------------------------------------------------------------
__global__ __launch_bounds__(256) void final_kernel(
    const float* __restrict__ inp, const float* __restrict__ ref,
    const int* __restrict__ curX, const int* __restrict__ curY,
    int* __restrict__ out_idx, float* __restrict__ out_s) {
  int p = blockIdx.x * blockDim.x + threadIdx.x;
  int x = curX[p], yy = curY[p];
  int q = yy * HW + x;
  out_idx[p] = q;
  float s = 0.f;
#pragma unroll 4
  for (int c = 0; c < NCH; ++c)
    s = fmaf(inp[(size_t)c * NPIX + p], ref[(size_t)c * NPIX + q], s);
  out_s[p] = s;
}

// ---------------------------------------------------------------------------
// host orchestration
// ---------------------------------------------------------------------------
extern "C" void kernel_launch(void* const* d_in, const int* in_sizes, int n_in,
                              void* d_out, int out_size, void* d_ws, size_t ws_size,
                              hipStream_t stream) {
  const float* input_map = (const float*)d_in[0];   // [192,128,128] f32
  const float* ref_map   = (const float*)d_in[1];   // [192,128,128] f32
  const float* inref_x   = (const float*)d_in[2];   // [1,1,128,128] f32
  const float* inref_y   = (const float*)d_in[3];   // [1,1,128,128] f32
  // scalars d_in[4..7] are fixed by setup: is_final=1, iters=2, minWH=127

  char*  ws  = (char*)d_ws;
  size_t off = 0;
  bf16_t* inB  = (bf16_t*)(ws + off); off += (size_t)NPIX * NCH * sizeof(bf16_t);
  bf16_t* refB = (bf16_t*)(ws + off); off += (size_t)NPIX * NCH * sizeof(bf16_t);
  int* bufX0 = (int*)(ws + off); off += (size_t)NPIX * sizeof(int);
  int* bufY0 = (int*)(ws + off); off += (size_t)NPIX * sizeof(int);
  int* bufX1 = (int*)(ws + off); off += (size_t)NPIX * sizeof(int);
  int* bufY1 = (int*)(ws + off); off += (size_t)NPIX * sizeof(int);

  const int packBlocks = (NCH * NPIX) / 256;
  pack_kernel<<<packBlocks, 256, 0, stream>>>(input_map, inB);
  pack_kernel<<<packBlocks, 256, 0, stream>>>(ref_map, refB);
  init_kernel<<<NPIX / 256, 256, 0, stream>>>(inref_x, inref_y, bufX0, bufY0);

  int *cx = bufX0, *cy = bufY0, *nx = bufX1, *ny = bufY1;
  for (int it = 0; it < 2; ++it) {
    for (int k = 1; k <= 127; k *= 2) {                 // propagation
      eval_kernel<0><<<HW, 256, 0, stream>>>(inB, refB, cx, cy, nx, ny, k, 0);
      int* t;
      t = cx; cx = nx; nx = t;
      t = cy; cy = ny; ny = t;
    }
    for (int k = 1; k <= 127; k *= 2) {                 // random search
      int m = it % k;
      eval_kernel<1><<<HW, 256, 0, stream>>>(inB, refB, cx, cy, nx, ny, k, m);
      int* t;
      t = cx; cx = nx; nx = t;
      t = cy; cy = ny; ny = t;
    }
  }

  final_kernel<<<NPIX / 256, 256, 0, stream>>>(input_map, ref_map, cx, cy,
                                               (int*)d_out,
                                               (float*)d_out + NPIX);
}
